// MultiHeadedAttentionAugv3_35940286333500
// MI455X (gfx1250) — matrix operand order repaired and checked by reading, not compiled, not run on hardware
//
#include <hip/hip_runtime.h>
#include <hip/hip_bf16.h>
#include <stdint.h>

#define DEV __device__ __forceinline__

typedef __bf16 bf16_t;
typedef __attribute__((ext_vector_type(16))) __bf16 bfrag;    // WMMA A/B operand (8 VGPRs)
typedef __attribute__((ext_vector_type(8)))  __bf16 bhalf8;   // 16-byte load unit
typedef __attribute__((ext_vector_type(8)))  float  f32x8;    // WMMA C/D operand
typedef __attribute__((ext_vector_type(4)))  unsigned int u32x4;
typedef __attribute__((ext_vector_type(8)))  int i32x8;
typedef __attribute__((ext_vector_type(4)))  int i32x4;

static constexpr int Bc  = 4;
static constexpr int Sc  = 1024;
static constexpr int DMc = 1024;
static constexpr int Hc  = 16;
static constexpr int DKc = 64;
static constexpr int Mrows = Bc * Sc;        // 4096 token rows
static constexpr int RH    = Bc * Hc * Sc;   // 65536 head-split rows
static constexpr int MW    = Sc / 32;        // packed mask words per row

#if defined(__HIP_DEVICE_COMPILE__) && defined(__gfx1250__) && __has_builtin(__builtin_amdgcn_tensor_load_to_lds)
#define HAVE_TDM 1
#else
#define HAVE_TDM 0
#endif

// ---------------- bf16 helpers ----------------
DEV bf16_t f2bf(float f) {
  uint32_t u = __builtin_bit_cast(uint32_t, f);
  uint32_t r = (u + 0x7FFFu + ((u >> 16) & 1u)) >> 16;   // round-to-nearest-even
  return __builtin_bit_cast(bf16_t, (uint16_t)r);
}

DEV bfrag make_frag(bhalf8 lo, bhalf8 hi) {
  bfrag r;
#pragma unroll
  for (int i = 0; i < 8; ++i) { r[i] = lo[i]; r[i + 8] = hi[i]; }
  return r;
}

// A-matrix fragment (16x32 bf16), ISA 7.12.2 layout:
// lane L: row = row_base + (L&15); elems 0..7 -> K = k_base + 8*hi + j,
//         elems 8..15 -> K = k_base + 16 + 8*hi + j   (hi = L>=16)
DEV bfrag load_a(const bf16_t* __restrict__ A, int lda, int row_base, int k_base, int lane) {
  const int hi = lane >> 4;
  const bf16_t* rp = A + (size_t)(row_base + (lane & 15)) * lda + k_base + 8 * hi;
  bhalf8 lo = *(const bhalf8*)rp;
  bhalf8 h2 = *(const bhalf8*)(rp + 16);
  return make_frag(lo, h2);
}

// B-matrix fragment (32x16 bf16) from row-major W[N][K] (B[k][n] = W[n][k]):
// lane L: col = col_base + (L&15); elems 0..7 -> K = k_base + 8*hi + j,
//         elems 8..15 -> K = k_base + 16 + 8*hi + j   (same as A)
DEV bfrag load_b(const bf16_t* __restrict__ W, int ldw, int col_base, int k_base, int lane) {
  const int hi = lane >> 4;
  const bf16_t* rp = W + (size_t)(col_base + (lane & 15)) * ldw + k_base + 8 * hi;
  bhalf8 lo = *(const bhalf8*)rp;
  bhalf8 h2 = *(const bhalf8*)(rp + 16);
  return make_frag(lo, h2);
}

DEV f32x8 wmma_bf16(bfrag a, bfrag b, f32x8 c) {
  return __builtin_amdgcn_wmma_f32_16x16x32_bf16(false, a, false, b, (short)0, c, false, false);
}

#if HAVE_TDM
// Issue a TDM 2-D tile load: rows x rowlen bf16 tile from a row-major
// (tensor_d1 x tensor_d0) bf16 tensor with row stride `stride_elems`,
// packed row-major into LDS at byte offset lds_off. D# per ISA 8.3/8.4.
// This toolchain exposes the 6-arg builtin: (g0, g1, g2, g3, extra, cpol).
DEV void tdm_load_2d(uint32_t lds_off, const bf16_t* gsrc,
                     uint32_t rows, uint32_t rowlen,
                     uint32_t stride_elems, uint32_t tensor_d0, uint32_t tensor_d1) {
  uint64_t ga = (uint64_t)(uintptr_t)gsrc;
  u32x4 g0;
  g0[0] = 1u;                                        // count=1, user descriptor
  g0[1] = lds_off;                                   // lds_addr (bytes)
  g0[2] = (uint32_t)ga;                              // global_addr[31:0]
  g0[3] = (uint32_t)((ga >> 32) & 0x01FFFFFFu) | 0x80000000u; // addr[56:32], type=2
  i32x8 g1;
  g1[0] = (int)(1u << 16);                           // data_size=1 (2 bytes); no multicast
  g1[1] = (int)((tensor_d0 & 0xFFFFu) << 16);        // tensor_dim0[15:0] @bits63:48
  g1[2] = (int)(((tensor_d0 >> 16) & 0xFFFFu) | ((tensor_d1 & 0xFFFFu) << 16));
  g1[3] = (int)(((tensor_d1 >> 16) & 0xFFFFu) | (rowlen << 16)); // tile_dim0
  g1[4] = (int)(rows & 0xFFFFu);                     // tile_dim1; tile_dim2=0 (2-D)
  g1[5] = (int)stride_elems;                         // tensor_dim0_stride[31:0]
  g1[6] = 0;
  g1[7] = 0;
  i32x4 z4 = {0, 0, 0, 0};                           // D# groups 2/3 unused (2-D tensor)
  i32x8 z8 = {0, 0, 0, 0, 0, 0, 0, 0};
  __builtin_amdgcn_tensor_load_to_lds(g0, g1, z4, z4, z8, 0);
}
DEV uint32_t lds_off(const void* p) { return (uint32_t)(uintptr_t)p; }
#endif

// ---------------- f32 -> bf16 conversion ----------------
__global__ void cvt_f32_bf16(const float* __restrict__ src, bf16_t* __restrict__ dst, int n) {
  int i = blockIdx.x * blockDim.x + threadIdx.x;
  int stride = gridDim.x * blockDim.x;
  for (; i < n; i += stride) dst[i] = f2bf(src[i]);
}

// ---------------- mask bit-packing (1 bit per key via wave ballot) ----------------
__global__ void pack_mask(const int* __restrict__ mask, uint32_t* __restrict__ pk, int n) {
  int i = blockIdx.x * blockDim.x + threadIdx.x;   // n is a multiple of blockDim
  if (i < n) {
    unsigned long long bal = __ballot(mask[i] != 0);
    if ((threadIdx.x & 31) == 0) pk[i >> 5] = (uint32_t)bal;
  }
}

// ---------------- generic 64x64-per-wave WMMA GEMM ----------------
// C[M,N] = A[M,K] @ W[N,K]^T + bias, optional ReLU.
// MODE 0: f32 row-major out.        MODE 1: bf16 head-split [B,H,S,DK].
// MODE 2: MODE1 + bf16 transposed [B,H,DK,S] to out2.   MODE 3: bf16 row-major.
template<int MODE, bool RELU>
__global__ __launch_bounds__(32)
void gemm_wave(const bf16_t* __restrict__ A, const bf16_t* __restrict__ W,
               const float* __restrict__ bias, void* __restrict__ outp,
               void* __restrict__ out2p, int M, int N, int K) {
  const int lane = threadIdx.x;
  const int col  = lane & 15;
  const int hi   = lane >> 4;
  const int nb   = blockIdx.x * 64;
  const int mb   = blockIdx.y * 64;

  f32x8 acc[4][4];
#pragma unroll
  for (int i = 0; i < 4; ++i)
#pragma unroll
    for (int j = 0; j < 4; ++j)
#pragma unroll
      for (int r = 0; r < 8; ++r) acc[i][j][r] = 0.0f;

#if HAVE_TDM
  // TDM double-buffered LDS staging: 64x32 bf16 tiles for A and W.
  __shared__ bf16_t smA[2][64 * 32];
  __shared__ bf16_t smB[2][64 * 32];
  tdm_load_2d(lds_off(&smA[0][0]), A + (size_t)mb * K, 64, 32, K, K, M);
  tdm_load_2d(lds_off(&smB[0][0]), W + (size_t)nb * K, 64, 32, K, K, N);
  int st = 0;
  for (int k = 0; k < K; k += 32, st ^= 1) {
    if (k + 32 < K) {
      // WAR: prior ds reads of the buffer we are about to refill must retire.
      asm volatile("s_wait_dscnt 0" ::: "memory");
      tdm_load_2d(lds_off(&smA[st ^ 1][0]), A + (size_t)mb * K + (k + 32), 64, 32, K, K, M);
      tdm_load_2d(lds_off(&smB[st ^ 1][0]), W + (size_t)nb * K + (k + 32), 64, 32, K, K, N);
      __builtin_amdgcn_s_wait_tensorcnt((short)2);   // current stage complete
    } else {
      __builtin_amdgcn_s_wait_tensorcnt((short)0);
    }
    asm volatile("" ::: "memory");
    bfrag af[4], bf_[4];
#pragma unroll
    for (int mi = 0; mi < 4; ++mi) af[mi]  = load_a(&smA[st][0], 32, 16 * mi, 0, lane);
#pragma unroll
    for (int ni = 0; ni < 4; ++ni) bf_[ni] = load_b(&smB[st][0], 32, 16 * ni, 0, lane);
#pragma unroll
    for (int mi = 0; mi < 4; ++mi)
#pragma unroll
      for (int ni = 0; ni < 4; ++ni)
        acc[mi][ni] = wmma_bf16(af[mi], bf_[ni], acc[mi][ni]);
  }
#else
  for (int k = 0; k < K; k += 32) {
    if (k + 256 < K) {   // global_prefetch_b8 a few K-steps ahead
      __builtin_prefetch(A + (size_t)(mb + col) * K + k + 256, 0, 3);
      __builtin_prefetch(W + (size_t)(nb + col) * K + k + 256, 0, 3);
    }
    bfrag af[4], bf_[4];
#pragma unroll
    for (int mi = 0; mi < 4; ++mi) af[mi]  = load_a(A, K, mb + 16 * mi, k, lane);
#pragma unroll
    for (int ni = 0; ni < 4; ++ni) bf_[ni] = load_b(W, K, nb + 16 * ni, k, lane);
#pragma unroll
    for (int mi = 0; mi < 4; ++mi)
#pragma unroll
      for (int ni = 0; ni < 4; ++ni)
        acc[mi][ni] = wmma_bf16(af[mi], bf_[ni], acc[mi][ni]);
  }
#endif

#pragma unroll
  for (int ni = 0; ni < 4; ++ni) {
    const int n = nb + 16 * ni + col;
    const float bv = bias ? bias[n] : 0.0f;
#pragma unroll
    for (int mi = 0; mi < 4; ++mi) {
#pragma unroll
      for (int r = 0; r < 8; ++r) {
        const int m = mb + 16 * mi + r + 8 * hi;
        float v = acc[mi][ni][r] + bv;
        if (RELU) v = fmaxf(v, 0.0f);
        if (MODE == 0) {
          ((float*)outp)[(size_t)m * N + n] = v;
        } else if (MODE == 3) {
          ((bf16_t*)outp)[(size_t)m * N + n] = f2bf(v);
        } else {  // MODE 1 / 2: head-split
          const int bidx = m >> 10, s = m & (Sc - 1);
          const int h = n >> 6, d = n & (DKc - 1);
          const size_t bh = (size_t)bidx * Hc + h;
          ((bf16_t*)outp)[(bh * Sc + s) * DKc + d] = f2bf(v);
          if (MODE == 2)
            ((bf16_t*)out2p)[(bh * DKc + d) * Sc + s] = f2bf(v);
        }
      }
    }
  }
}

// ---------------- fused dual-score flash attention ----------------
// kq = q1@kr^T, qv = q2@vr^T ; attn = softmax(mask(kq)) ; out = (attn*qv)@K
// One wave32 per 16-row query tile; streams 32-key chunks (online softmax).
__global__ __launch_bounds__(32)
void attn_kernel(const bf16_t* __restrict__ q1, const bf16_t* __restrict__ q2,
                 const bf16_t* __restrict__ kr, const bf16_t* __restrict__ vr,
                 const bf16_t* __restrict__ khT, const uint32_t* __restrict__ maskp,
                 bf16_t* __restrict__ attn_out) {
  __shared__ bf16_t lds[16 * 32];      // P-tile bounce: C-layout -> A-layout
  const int lane = threadIdx.x;
  const int col  = lane & 15;
  const int hi   = lane >> 4;
  const int qt   = blockIdx.x;         // query tile: 0..63
  const int bh   = blockIdx.y;         // (b*H + h): 0..63
  const int b    = bh >> 4;
  const int h    = bh & 15;

  const bf16_t* q1p = q1  + ((size_t)bh * Sc + qt * 16) * DKc;
  const bf16_t* q2p = q2  + ((size_t)bh * Sc + qt * 16) * DKc;
  const bf16_t* krp = kr  + (size_t)bh * Sc * DKc;
  const bf16_t* vrp = vr  + (size_t)bh * Sc * DKc;
  const bf16_t* ktp = khT + (size_t)bh * DKc * Sc;
  const uint32_t* mrow = maskp + ((size_t)b * Sc + qt * 16) * MW;

  bfrag aq1[2], aq2[2];
#pragma unroll
  for (int kc = 0; kc < 2; ++kc) {
    aq1[kc] = load_a(q1p, DKc, 0, 32 * kc, lane);
    aq2[kc] = load_a(q2p, DKc, 0, 32 * kc, lane);
  }

  f32x8 oacc[4];
#pragma unroll
  for (int t = 0; t < 4; ++t)
#pragma unroll
    for (int r = 0; r < 8; ++r) oacc[t][r] = 0.0f;
  float m_[8], l_[8];
#pragma unroll
  for (int r = 0; r < 8; ++r) { m_[r] = -3.0e38f; l_[r] = 0.0f; }

  for (int kb = 0; kb < Sc / 32; ++kb) {
    const int key0 = kb * 32;

    // ---- score tiles: kq (q1.kr) and qv (q2.vr), 16x32 each
    f32x8 skq[2], sqv[2];
#pragma unroll
    for (int t = 0; t < 2; ++t) {
#pragma unroll
      for (int r = 0; r < 8; ++r) { skq[t][r] = 0.0f; sqv[t][r] = 0.0f; }
#pragma unroll
      for (int kc = 0; kc < 2; ++kc) {
        bfrag bk = load_b(krp, DKc, key0 + 16 * t, 32 * kc, lane);
        skq[t] = wmma_bf16(aq1[kc], bk, skq[t]);
        bfrag bv = load_b(vrp, DKc, key0 + 16 * t, 32 * kc, lane);
        sqv[t] = wmma_bf16(aq2[kc], bv, sqv[t]);
      }
    }

    // ---- bit-packed mask + online softmax (rows at r+8*hi; cols across lanes)
    float s0[8], s1[8], corr_[8];
#pragma unroll
    for (int r = 0; r < 8; ++r) {
      const uint32_t mw = mrow[(size_t)(r + 8 * hi) * MW + kb];
      s0[r] = ((mw >> col) & 1u)        ? skq[0][r] : -1.0e9f;
      s1[r] = ((mw >> (col + 16)) & 1u) ? skq[1][r] : -1.0e9f;
    }
#pragma unroll
    for (int r = 0; r < 8; ++r) {
      float t = fmaxf(s0[r], s1[r]);
#pragma unroll
      for (int x = 1; x < 16; x <<= 1) t = fmaxf(t, __shfl_xor(t, x, 32));
      const float mn = fmaxf(m_[r], t);
      corr_[r] = __expf(m_[r] - mn);
      const float p0 = __expf(s0[r] - mn);
      const float p1 = __expf(s1[r] - mn);
      float rs = p0 + p1;
#pragma unroll
      for (int x = 1; x < 16; x <<= 1) rs += __shfl_xor(rs, x, 32);
      l_[r] = l_[r] * corr_[r] + rs;
      m_[r] = mn;
      s0[r] = p0 * sqv[0][r];   // P = exp(kq-m) * qv  (pre-normalization)
      s1[r] = p1 * sqv[1][r];
    }
#pragma unroll
    for (int t = 0; t < 4; ++t)
#pragma unroll
      for (int r = 0; r < 8; ++r) oacc[t][r] *= corr_[r];

    // ---- transpose P (C-layout) -> A-layout via LDS
#pragma unroll
    for (int r = 0; r < 8; ++r) {
      lds[(r + 8 * hi) * 32 + col]      = f2bf(s0[r]);
      lds[(r + 8 * hi) * 32 + 16 + col] = f2bf(s1[r]);
    }
    __syncthreads();
    bfrag pA;
    {
      const bf16_t* lp = lds + col * 32 + 8 * hi;
      bhalf8 lo = *(const bhalf8*)lp;
      bhalf8 h2 = *(const bhalf8*)(lp + 16);
      pA = make_frag(lo, h2);
    }
    __syncthreads();

    // ---- out(16x64) += P(16x32) @ K_chunk(32x64); K fed from khT[d][s]
#pragma unroll
    for (int t = 0; t < 4; ++t) {
      bfrag bk = load_b(ktp, Sc, 16 * t, key0, lane);
      oacc[t] = wmma_bf16(pA, bk, oacc[t]);
    }
  }

  // ---- normalize and write merged [B,S,DM] bf16
  bf16_t* op = attn_out + ((size_t)(b * Sc + qt * 16)) * DMc + h * DKc;
#pragma unroll
  for (int t = 0; t < 4; ++t)
#pragma unroll
    for (int r = 0; r < 8; ++r) {
      const float v = oacc[t][r] / l_[r];
      op[(size_t)(r + 8 * hi) * DMc + 16 * t + col] = f2bf(v);
    }
}

// ---------------- host orchestration ----------------
extern "C" void kernel_launch(void* const* d_in, const int* in_sizes, int n_in,
                              void* d_out, int out_size, void* d_ws, size_t ws_size,
                              hipStream_t stream) {
  const float* query = (const float*)d_in[0];
  const float* key   = (const float*)d_in[1];
  const float* value = (const float*)d_in[2];
  const int*   mask  = (const int*)  d_in[3];
  const float* Wq  = (const float*)d_in[4],  *bq  = (const float*)d_in[5];
  const float* Wk  = (const float*)d_in[6],  *bk  = (const float*)d_in[7];
  const float* Wv  = (const float*)d_in[8],  *bv  = (const float*)d_in[9];
  const float* Wo  = (const float*)d_in[10], *bo  = (const float*)d_in[11];
  const float* Wkl = (const float*)d_in[12], *bkl = (const float*)d_in[13];
  const float* Wql = (const float*)d_in[14], *bql = (const float*)d_in[15];
  const float* Wq2 = (const float*)d_in[16], *bq2 = (const float*)d_in[17];
  const float* Wvl = (const float*)d_in[18], *bvl = (const float*)d_in[19];
  const float* Wel = (const float*)d_in[20], *bel = (const float*)d_in[21];

  char* ws = (char*)d_ws;
  size_t off = 0;
  auto alloc = [&](size_t bytes) -> char* {
    char* p = ws + off;
    off += (bytes + 255) & ~(size_t)255;
    return p;
  };
  const size_t TOK = (size_t)Mrows * DMc;       // 4M elems
  const size_t HSZ = (size_t)RH * DKc;          // 4M elems (head-split)

  bf16_t* xq   = (bf16_t*)alloc(TOK * 2);
  bf16_t* xk   = (bf16_t*)alloc(TOK * 2);
  bf16_t* xv   = (bf16_t*)alloc(TOK * 2);
  bf16_t* Wq_b = (bf16_t*)alloc((size_t)DMc * DMc * 2);
  bf16_t* Wk_b = (bf16_t*)alloc((size_t)DMc * DMc * 2);
  bf16_t* Wv_b = (bf16_t*)alloc((size_t)DMc * DMc * 2);
  bf16_t* Wo_b = (bf16_t*)alloc((size_t)DMc * DMc * 2);
  bf16_t* Wkl_b = (bf16_t*)alloc((size_t)DKc * DKc * 2);
  bf16_t* Wql_b = (bf16_t*)alloc((size_t)DKc * DKc * 2);
  bf16_t* Wq2_b = (bf16_t*)alloc((size_t)DKc * DKc * 2);
  bf16_t* Wvl_b = (bf16_t*)alloc((size_t)DKc * DKc * 2);
  bf16_t* Wel_b = (bf16_t*)alloc((size_t)DKc * DKc * 2);
  bf16_t* qh   = (bf16_t*)alloc(HSZ * 2);
  bf16_t* kh   = (bf16_t*)alloc(HSZ * 2);
  bf16_t* vh   = (bf16_t*)alloc(HSZ * 2);
  bf16_t* khT  = (bf16_t*)alloc(HSZ * 2);
  bf16_t* q2b  = (bf16_t*)alloc(HSZ * 2);
  bf16_t* krb  = (bf16_t*)alloc(HSZ * 2);
  bf16_t* vrb  = (bf16_t*)alloc(HSZ * 2);
  uint32_t* mpk = (uint32_t*)alloc((size_t)Bc * Sc * MW * 4);  // bit-packed mask
  // alias buffers that are dead by the time these are written:
  bf16_t* tmpb     = xq;   // q1 intermediate (xq dead after Q projection)
  bf16_t* q1b      = xv;   // xv dead after V projection
  bf16_t* attn_out = xk;   // xk dead after K projection

  auto cvt = [&](const float* s, bf16_t* d, size_t n) {
    int blocks = (int)((n + 1023) / 1024);
    if (blocks > 2048) blocks = 2048;
    cvt_f32_bf16<<<dim3(blocks), dim3(256), 0, stream>>>(s, d, (int)n);
  };
  cvt(query, xq, TOK);
  cvt(key,   xk, TOK);
  cvt(value, xv, TOK);
  cvt(Wq,  Wq_b,  (size_t)DMc * DMc);
  cvt(Wk,  Wk_b,  (size_t)DMc * DMc);
  cvt(Wv,  Wv_b,  (size_t)DMc * DMc);
  cvt(Wo,  Wo_b,  (size_t)DMc * DMc);
  cvt(Wkl, Wkl_b, (size_t)DKc * DKc);
  cvt(Wql, Wql_b, (size_t)DKc * DKc);
  cvt(Wq2, Wq2_b, (size_t)DKc * DKc);
  cvt(Wvl, Wvl_b, (size_t)DKc * DKc);
  cvt(Wel, Wel_b, (size_t)DKc * DKc);

  {
    const int nmask = Bc * Sc * Sc;
    pack_mask<<<dim3(nmask / 256), dim3(256), 0, stream>>>(mask, mpk, nmask);
  }

  // Q/K/V projections (head-split; K also transposed for the PV matmul)
  const dim3 gProj(DMc / 64, Mrows / 64);   // (16, 64)
  gemm_wave<1, false><<<gProj, 32, 0, stream>>>(xq, Wq_b, bq, qh, nullptr, Mrows, DMc, DMc);
  gemm_wave<2, false><<<gProj, 32, 0, stream>>>(xk, Wk_b, bk, kh, khT,     Mrows, DMc, DMc);
  gemm_wave<1, false><<<gProj, 32, 0, stream>>>(xv, Wv_b, bv, vh, nullptr, Mrows, DMc, DMc);

  // per-head DKxDK refinement linears (+ReLU), all heads share the weights
  const dim3 gRef(1, RH / 64);              // (1, 1024)
  gemm_wave<3, true><<<gRef, 32, 0, stream>>>(kh,   Wkl_b, bkl, krb,  nullptr, RH, DKc, DKc);
  gemm_wave<3, true><<<gRef, 32, 0, stream>>>(qh,   Wql_b, bql, tmpb, nullptr, RH, DKc, DKc);
  gemm_wave<3, true><<<gRef, 32, 0, stream>>>(tmpb, Wel_b, bel, q1b,  nullptr, RH, DKc, DKc);
  gemm_wave<3, true><<<gRef, 32, 0, stream>>>(qh,   Wq2_b, bq2, q2b,  nullptr, RH, DKc, DKc);
  gemm_wave<3, true><<<gRef, 32, 0, stream>>>(vh,   Wvl_b, bvl, vrb,  nullptr, RH, DKc, DKc);

  // fused dual-score attention -> merged [B,S,DM] bf16
  attn_kernel<<<dim3(Sc / 16, Bc * Hc), 32, 0, stream>>>(q1b, q2b, krb, vrb, khT, mpk, attn_out);

  // final output projection, f32 result
  gemm_wave<0, false><<<gProj, 32, 0, stream>>>(attn_out, Wo_b, bo, d_out, nullptr, Mrows, DMc, DMc);
}
